// PNAConvTower_9826885173949
// MI455X (gfx1250) — compile-verified, matrix operand order checked by reading, not run on hardware
//
#include <hip/hip_runtime.h>
#include <hip/hip_bf16.h>
#include <math.h>

// PNA conv for MI455X (gfx1250, wave32).
// msg factors into phi_src/phi_dst node GEMMs + per-edge adds (L2-resident
// scatter). GEMMs: v_wmma_f32_16x16x32_f16 with fragment-major LDS operands
// so the hot loop uses ds_load_b128 instead of 16x ds_load_u16 per fragment.

#define NN    50000
#define EE    800000
#define FIN   64
#define FOUT  64
#define KTOT  832            // 13*64
#define NKT   26             // KTOT/32
#define PNA_DELTA 2.83f
#define MOM_EPS 1e-30f
#define BN_EPS  1e-5f

typedef _Float16 v16h __attribute__((ext_vector_type(16)));
typedef float    v8f  __attribute__((ext_vector_type(8)));

static __device__ __forceinline__ v8f wmma16(v16h a, v16h b, v8f c) {
  return __builtin_amdgcn_wmma_f32_16x16x32_f16(false, a, false, b, (short)0, c, false, false);
}

// fragment-major swizzle: element j of lane ln within a 32-K-slice.
// lane = (row_or_col & 15) + 16*h ; kl -> (h, j):
//   kl in [0,8):h=0,j=kl  [8,16):h=1,j=kl-8  [16,24):h=0,j=8+kl-16  [24,32):h=1,j=8+kl-24
static __device__ __forceinline__ int frag_h(int kl) { return (kl >> 3) & 1; }
static __device__ __forceinline__ int frag_j(int kl) { return ((kl & 16) ? 8 : 0) + (kl & 7); }

// order-preserving float<->uint keys so max/min use integer atomics
static __device__ __forceinline__ unsigned fkey(float v) {
  unsigned u = __float_as_uint(v);
  return (u & 0x80000000u) ? ~u : (u | 0x80000000u);
}
static __device__ __forceinline__ float fdec(unsigned k) {
  unsigned u = (k & 0x80000000u) ? (k & 0x7fffffffu) : ~k;
  return __uint_as_float(u);
}

// ---------------- kernel 0: init scratch ----------------
__global__ void k_init(float* __restrict__ s_sum, float* __restrict__ s_sq,
                       unsigned* __restrict__ s_max, unsigned* __restrict__ s_min,
                       unsigned* __restrict__ deg, float* __restrict__ bn) {
  int i = blockIdx.x * 256 + threadIdx.x;
  if (i < NN * FIN) {
    s_sum[i] = 0.f;
    s_sq[i]  = 0.f;
    s_max[i] = 0u;
    s_min[i] = 0xFFFFFFFFu;
  }
  if (i < NN) deg[i] = 0u;
  if (i < 2 * FOUT) bn[i] = 0.f;
}

// ---------------- kernel 1: phi GEMMs ----------------
// phi_src = feat @ Wm[0:64,:], phi_dst = feat @ Wm[64:128,:] + bm
__global__ void k_phi(const float* __restrict__ feat, const float* __restrict__ Wm,
                      const float* __restrict__ bm,
                      float* __restrict__ phi_s, float* __restrict__ phi_d) {
  __shared__ v16h WmF[16 * 32];          // 16 fragments (kt 0..3 x w 0..3)
  _Float16* WmL = (_Float16*)WmF;
  int tid = threadIdx.x;
  for (int i = tid; i < 128 * 64; i += 128) {   // coalesced global read
    int k = i >> 6, c = i & 63;
    int kt = k >> 5, kl = k & 31;
    int ln = (c & 15) + 16 * frag_h(kl);
    WmL[((kt * 4 + (c >> 4)) * 32 + ln) * 16 + frag_j(kl)] = (_Float16)Wm[i];
  }
  __syncthreads();

  int lane = tid & 31, wave = tid >> 5;
  int rr = lane & 15, hh = lane >> 4;
  int row0 = blockIdx.x * 16;
  int cn = wave * 16 + rr;

  v8f accS = {}; v8f accD = {};
  #pragma unroll
  for (int k0 = 0; k0 < 64; k0 += 32) {
    v16h a;
    {
      const float* p = feat + (size_t)(row0 + rr) * 64 + k0 + 8 * hh;
      float4 u0 = *(const float4*)(p);
      float4 u1 = *(const float4*)(p + 4);
      float4 u2 = *(const float4*)(p + 16);
      float4 u3 = *(const float4*)(p + 20);
      a[0]=(_Float16)u0.x; a[1]=(_Float16)u0.y; a[2]=(_Float16)u0.z; a[3]=(_Float16)u0.w;
      a[4]=(_Float16)u1.x; a[5]=(_Float16)u1.y; a[6]=(_Float16)u1.z; a[7]=(_Float16)u1.w;
      a[8]=(_Float16)u2.x; a[9]=(_Float16)u2.y; a[10]=(_Float16)u2.z; a[11]=(_Float16)u2.w;
      a[12]=(_Float16)u3.x; a[13]=(_Float16)u3.y; a[14]=(_Float16)u3.z; a[15]=(_Float16)u3.w;
    }
    int kt = k0 >> 5;
    v16h bsrc = WmF[((kt)     * 4 + wave) * 32 + lane];   // rows k0..k0+31
    v16h bdst = WmF[((kt + 2) * 4 + wave) * 32 + lane];   // rows 64+k0..
    accS = wmma16(a, bsrc, accS);
    accD = wmma16(a, bdst, accD);
  }
  float bmv = bm[cn];
  #pragma unroll
  for (int v = 0; v < 8; ++v) {
    int m = v + 8 * hh;
    size_t o = (size_t)(row0 + m) * 64 + cn;
    phi_s[o] = accS[v];
    phi_d[o] = accD[v] + bmv;
  }
}

// ---------------- kernel 2: edge scatter ----------------
__global__ void k_edge(const int* __restrict__ src, const int* __restrict__ dst,
                       const float* __restrict__ phi_s, const float* __restrict__ phi_d,
                       float* __restrict__ s_sum, float* __restrict__ s_sq,
                       unsigned* __restrict__ s_max, unsigned* __restrict__ s_min,
                       unsigned* __restrict__ deg) {
  int wave = threadIdx.x >> 5, lane = threadIdx.x & 31;
  int e = blockIdx.x * 8 + wave;
  int s = src[e], d = dst[e];
  int f = lane * 2;
  float2 ps = *(const float2*)(phi_s + (size_t)s * 64 + f);
  float2 pd = *(const float2*)(phi_d + (size_t)d * 64 + f);
  float m0 = ps.x + pd.x;
  float m1 = ps.y + pd.y;
  size_t base = (size_t)d * 64 + f;
  atomicAdd(&s_sum[base],     m0);
  atomicAdd(&s_sum[base + 1], m1);
  atomicAdd(&s_sq[base],      m0 * m0);
  atomicAdd(&s_sq[base + 1],  m1 * m1);
  atomicMax(&s_max[base],     fkey(m0));
  atomicMax(&s_max[base + 1], fkey(m1));
  atomicMin(&s_min[base],     fkey(m0));
  atomicMin(&s_min[base + 1], fkey(m1));
  if (lane == 0) atomicAdd(&deg[d], 1u);
}

// ---------------- kernel 3: node update GEMM ----------------
// z = [feat | masked(h, h*a, h*b)] @ Wu + bu, * 1/sqrt(N); fused BN stats.
// 625 blocks x 128 threads; Wu staged fragment-major once per block; 5 tiles.
__global__ void k_node(const float* __restrict__ feat, const float* __restrict__ Wu,
                       const float* __restrict__ bu,
                       const float* __restrict__ s_sum, const float* __restrict__ s_sq,
                       const unsigned* __restrict__ s_max, const unsigned* __restrict__ s_min,
                       const unsigned* __restrict__ deg,
                       float* __restrict__ zbuf, float* __restrict__ bn) {
  extern __shared__ v16h smv[];
  v16h* WuF = smv;                      // 26*4 = 104 fragments
  v16h* XF  = smv + (NKT * 4) * 32;     // 26 fragments (one per k-tile)
  _Float16* WuL = (_Float16*)WuF;
  _Float16* XL  = (_Float16*)XF;
  int tid = threadIdx.x;
  for (int i = tid; i < KTOT * FOUT; i += 128) {   // coalesced global read
    int k = i >> 6, c = i & 63;
    int kt = k >> 5, kl = k & 31;
    int ln = (c & 15) + 16 * frag_h(kl);
    WuL[((kt * 4 + (c >> 4)) * 32 + ln) * 16 + frag_j(kl)] = (_Float16)Wu[i];
  }
  int lane = tid & 31, wave = tid >> 5;
  int rr = lane & 15, hh = lane >> 4;
  int cn = wave * 16 + rr;
  const float snorm = 4.4721360e-3f;    // 1/sqrt(50000)

  for (int t = blockIdx.x; t < NN / 16; t += gridDim.x) {
    int row0 = t * 16;
    __syncthreads();                    // prior compute done (and Wu staged)
    {                                   // stage X panel: thread -> (row, 8 feats)
      int r = tid >> 3;
      int f0 = (tid & 7) * 8;
      int node = row0 + r;
      float D = (float)deg[node];
      bool has = D > 0.f;
      float sd = fmaxf(D, 1.f);
      float lg = logf(D + 1.f);
      float slg = (lg > 0.f) ? lg : 1.f;
      float as = slg * (1.f / PNA_DELTA);
      float bs = PNA_DELTA / slg;
      #pragma unroll
      for (int u = 0; u < 8; ++u) {
        int f = f0 + u;
        size_t idx = (size_t)node * 64 + f;
        float mean = s_sum[idx] / sd;
        float msq  = s_sq[idx] / sd;
        float var  = fmaxf(msq - mean * mean, 0.f);
        float stdv = sqrtf(var + MOM_EPS);
        float mx   = has ? fdec(s_max[idx]) : 0.f;
        float mn   = has ? fdec(s_min[idx]) : 0.f;
        if (!has) { mean = 0.f; stdv = 0.f; }
        float vals[13] = { feat[idx],
                           mean, mx, mn, stdv,
                           mean * as, mx * as, mn * as, stdv * as,
                           mean * bs, mx * bs, mn * bs, stdv * bs };
        #pragma unroll
        for (int s13 = 0; s13 < 13; ++s13) {
          int kcol = s13 * 64 + f;
          int kt = kcol >> 5, kl = kcol & 31;
          int ln = r + 16 * frag_h(kl);
          XL[(kt * 32 + ln) * 16 + frag_j(kl)] = (_Float16)vals[s13];
        }
      }
    }
    __syncthreads();

    v8f acc = {};
    #pragma unroll 2
    for (int kt = 0; kt < NKT; ++kt) {
      v16h a = XF[kt * 32 + lane];                 // 2x ds_load_b128
      v16h b = WuF[(kt * 4 + wave) * 32 + lane];   // 2x ds_load_b128
      acc = wmma16(a, b, acc);
    }
    float buv = bu[cn];
    float psum = 0.f, psq = 0.f;
    #pragma unroll
    for (int v = 0; v < 8; ++v) {
      int m = v + 8 * hh;
      float zv = (acc[v] + buv) * snorm;
      zbuf[(size_t)(row0 + m) * 64 + cn] = zv;
      psum += zv; psq += zv * zv;
    }
    atomicAdd(&bn[cn], psum);
    atomicAdd(&bn[64 + cn], psq);
  }
}

// ---------------- kernel 4: batchnorm finalize ----------------
__global__ void k_bn(const float* __restrict__ zbuf, const float* __restrict__ bn,
                     const float* __restrict__ gamma, const float* __restrict__ beta,
                     float* __restrict__ out) {
  int i = blockIdx.x * 256 + threadIdx.x;
  if (i >= NN * FOUT) return;
  int c = i & 63;
  const float invN = 1.f / (float)NN;
  float mu = bn[c] * invN;
  float var = fmaxf(bn[64 + c] * invN - mu * mu, 0.f);
  float rstd = rsqrtf(var + BN_EPS);
  out[i] = (zbuf[i] - mu) * rstd * gamma[c] + beta[c];
}

extern "C" void kernel_launch(void* const* d_in, const int* in_sizes, int n_in,
                              void* d_out, int out_size, void* d_ws, size_t ws_size,
                              hipStream_t stream) {
  (void)in_sizes; (void)n_in; (void)out_size; (void)ws_size;
  const float* feat  = (const float*)d_in[0];
  const float* Wm    = (const float*)d_in[1];
  const float* bm    = (const float*)d_in[2];
  const float* Wu    = (const float*)d_in[3];
  const float* bu    = (const float*)d_in[4];
  const float* gamma = (const float*)d_in[5];
  const float* beta  = (const float*)d_in[6];
  const int*   src   = (const int*)d_in[7];
  const int*   dst   = (const int*)d_in[8];
  float* out = (float*)d_out;

  float* ws      = (float*)d_ws;
  float* phi_s   = ws;                          // N*64
  float* phi_d   = phi_s + (size_t)NN * FIN;    // N*64
  float* s_sum   = phi_d + (size_t)NN * FIN;    // N*64
  float* s_sq    = s_sum + (size_t)NN * FIN;    // N*64
  unsigned* s_max = (unsigned*)(s_sq + (size_t)NN * FIN);  // N*64
  unsigned* s_min = s_max + (size_t)NN * FIN;   // N*64
  unsigned* deg   = s_min + (size_t)NN * FIN;   // N
  float* zbuf    = (float*)(deg + NN);          // N*64
  float* bn      = zbuf + (size_t)NN * FOUT;    // 128

  k_init<<<(NN * FIN + 255) / 256, 256, 0, stream>>>(s_sum, s_sq, s_max, s_min, deg, bn);
  k_phi<<<NN / 16, 128, 0, stream>>>(feat, Wm, bm, phi_s, phi_d);
  k_edge<<<EE / 8, 256, 0, stream>>>(src, dst, phi_s, phi_d, s_sum, s_sq, s_max, s_min, deg);
  size_t lds_bytes = (size_t)(NKT * 4 * 32 + NKT * 32) * sizeof(v16h); // 133120 B
  k_node<<<625, 128, lds_bytes, stream>>>(feat, Wu, bu, s_sum, s_sq, s_max, s_min, deg, zbuf, bn);
  k_bn<<<(NN * FOUT + 255) / 256, 256, 0, stream>>>(zbuf, bn, gamma, beta, out);
}
